// SA_Layer_8091718386271
// MI455X (gfx1250) — matrix operand order repaired and checked by reading, not compiled, not run on hardware
//
#include <hip/hip_runtime.h>
#include <stdint.h>

// ---------------- problem constants ----------------
#define B_    8
#define P_    8192
#define C_    64
#define M_    2048            // P/4
#define KNN_  32
#define L_    (B_*M_*KNN_)    // 524288 rows through the MLP
#define LDA1  96              // 67 padded to 3*32 for bf16 WMMA K-chunks
#define CENT_OFF (B_*M_*3)    // centers come first in d_out

typedef __attribute__((ext_vector_type(16))) __bf16 v16bf;
typedef __attribute__((ext_vector_type(8)))  float  v8f;
typedef int v4i __attribute__((vector_size(16)));   // matches builtin's V4i

#define GLOBAL_AS __attribute__((address_space(1)))
#define LDS_AS    __attribute__((address_space(3)))

__device__ __forceinline__ unsigned short f2bf(float f) {
  union { float f; uint32_t u; } x; x.f = f;
  uint32_t u = x.u;
  u += 0x7FFFu + ((u >> 16) & 1u);     // round-to-nearest-even
  return (unsigned short)(u >> 16);
}

// ---------------------------------------------------------------------------
// Kernel 1: per-center KNN + gather into padded bf16 activation matrix.
// One workgroup per (b, m). d^2 for all 8192 points lives in LDS (32 KB of
// the 320 KB WGP LDS); top-32 selected by 32 rounds of tree argmin.
// ---------------------------------------------------------------------------
__global__ __launch_bounds__(256) void knn_gather_kernel(
    const float* __restrict__ xyz, const float* __restrict__ feats,
    unsigned short* __restrict__ cat, float* __restrict__ outCenters)
{
  __shared__ float sd[P_];
  __shared__ float rmin[256];
  __shared__ int   ridx[256];
  __shared__ int   knn[KNN_];
  __shared__ float ctr[3];

  const int g   = blockIdx.x;          // b*M + m
  const int b   = g / M_;
  const int m   = g - b * M_;
  const int tid = threadIdx.x;

  if (tid == 0) {
    const float step = (float)(P_ - 1) / (float)(M_ - 1);   // linspace step
    int idxc = (int)(step * (float)m);
    if (idxc > P_ - 1) idxc = P_ - 1;
    const float* cp = xyz + ((size_t)b * P_ + idxc) * 3;
    ctr[0] = cp[0]; ctr[1] = cp[1]; ctr[2] = cp[2];
    float* oc = outCenters + (size_t)g * 3;
    oc[0] = cp[0]; oc[1] = cp[1]; oc[2] = cp[2];
  }
  __syncthreads();

  const float cx = ctr[0], cy = ctr[1], cz = ctr[2];
  const float* xb = xyz + (size_t)b * P_ * 3;
  for (int p = tid; p < P_; p += 256) {
    float dx = xb[p*3+0] - cx, dy = xb[p*3+1] - cy, dz = xb[p*3+2] - cz;
    sd[p] = dx*dx + dy*dy + dz*dz;
  }
  __syncthreads();

  for (int it = 0; it < KNN_; ++it) {
    float best = 3.4e38f; int bi = P_;
    for (int p = tid; p < P_; p += 256) {
      float v = sd[p];
      if (v < best) { best = v; bi = p; }   // strict '<' => lowest index wins
    }
    rmin[tid] = best; ridx[tid] = bi;
    __syncthreads();
    for (int s = 128; s > 0; s >>= 1) {
      if (tid < s) {
        float o = rmin[tid+s]; int oi = ridx[tid+s];
        if (o < rmin[tid] || (o == rmin[tid] && oi < ridx[tid])) {
          rmin[tid] = o; ridx[tid] = oi;
        }
      }
      __syncthreads();
    }
    if (tid == 0) { knn[it] = ridx[0]; sd[ridx[0]] = 3.4e38f; }
    __syncthreads();
  }

  // Gather [local_xyz(3) | feats(64) | zero-pad(29)] rows as bf16.
  const float* fb = feats + (size_t)b * C_ * P_;
  unsigned short* crow = cat + (size_t)g * KNN_ * LDA1;
  for (int e = tid; e < KNN_ * LDA1; e += 256) {
    int j = e / LDA1, col = e - j * LDA1;
    int p = knn[j];
    float val;
    if (col < 3)       val = xb[p*3 + col] - ctr[col];
    else if (col < 67) val = fb[(size_t)(col - 3) * P_ + p];
    else               val = 0.f;
    crow[(size_t)j * LDA1 + col] = f2bf(val);
  }
}

// ---------------------------------------------------------------------------
// Kernel 2: pack w (O x I, row-major fp32) into WMMA B-fragment layout
// (16x16x32 bf16, ISA 7.12.2): lane n = lane&15, K = (v>=4?16:0) + (lane>>4)*8 + 2*(v&3) + {0,1}.
// Flat index: ((kc*Nt + nt)*32 + lane)*8 + v, each entry = packed bf16 pair.
// ---------------------------------------------------------------------------
__global__ void pack_w_kernel(const float* __restrict__ w, int O, int I,
                              int Kc, int Nt, uint32_t* __restrict__ outp)
{
  int total = Kc * Nt * 256;
  for (int e = blockIdx.x * blockDim.x + threadIdx.x; e < total;
       e += gridDim.x * blockDim.x) {
    int v    = e & 7;
    int lane = (e >> 3) & 31;
    int nt   = (e >> 8) % Nt;
    int kc   = (e >> 8) / Nt;
    int n  = nt * 16 + (lane & 15);
    int k0 = kc * 32 + ((v & 4) << 2) + ((lane >> 4) * 8) + 2 * (v & 3);
    float a = (n < O && k0     < I) ? w[(size_t)n * I + k0]     : 0.f;
    float b = (n < O && k0 + 1 < I) ? w[(size_t)n * I + k0 + 1] : 0.f;
    outp[e] = (uint32_t)f2bf(a) | ((uint32_t)f2bf(b) << 16);
  }
}

// ---------------------------------------------------------------------------
// Kernel 3: bf16 WMMA GEMM + bias, emitting y (fp32) and deterministic
// per-block (sum, sumsq) partials per channel for the global batch-norm.
// 8 waves per block; each wave owns a 16-row tile and all NT column tiles.
// Weight fragments staged in LDS via async global->LDS DMA (ASYNCcnt path).
// LDA = KC*32 and N = NT*16 are compile-time so all hot-loop addresses fold
// into immediate offsets (no 64-bit multiplies in the loop).
// A-fragment layout (16-bit A 16x32, ISA 7.12.2):
//   lane: m = lane&15, h = lane>>4 ; VGPR v: k = (v>=4?16:0) + h*8 + 2*(v&3).
// C/D layout: VGPR r -> row r + 8*(lane>>4), col = lane&15 (per n-tile).
// ---------------------------------------------------------------------------
template<int KC, int NT>
__global__ __launch_bounds__(256) void gemm_bn_kernel(
    const unsigned short* __restrict__ A,
    const uint32_t* __restrict__ Bpack,
    const float* __restrict__ bias,
    float* __restrict__ Y,
    float* __restrict__ part, int ntiles)
{
  constexpr int LDA = KC * 32;   // A row stride (bf16 elements)
  constexpr int N   = NT * 16;   // output channels == Y row stride
  __shared__ uint32_t sB[KC * NT * 256];
  __shared__ float pSum[8 * 256];
  __shared__ float pSqr[8 * 256];

  const int tid  = threadIdx.x;
  const int lane = tid & 31;
  const int wave = tid >> 5;

  // Stage weight fragments to LDS with the gfx1250 async global->LDS path.
#if __has_builtin(__builtin_amdgcn_global_load_async_to_lds_b128) && \
    __has_builtin(__builtin_amdgcn_s_wait_asynccnt)
  for (int i = tid * 4; i < KC * NT * 256; i += 1024) {
    __builtin_amdgcn_global_load_async_to_lds_b128(
        (GLOBAL_AS v4i*)(Bpack + i),
        (LDS_AS v4i*)(&sB[i]), 0, 0);
  }
  __builtin_amdgcn_s_wait_asynccnt(0);
#else
  for (int i = tid; i < KC * NT * 256; i += 256) sB[i] = Bpack[i];
#endif
  __syncthreads();

  float bv[NT];
  #pragma unroll
  for (int nt = 0; nt < NT; ++nt) bv[nt] = bias[nt * 16 + (lane & 15)];

  float ssum[NT], ssq[NT];
  #pragma unroll
  for (int nt = 0; nt < NT; ++nt) { ssum[nt] = 0.f; ssq[nt] = 0.f; }

  const int mrow = lane & 15;
  const int h    = lane >> 4;
  const int totalWaves = gridDim.x * 8;

  for (int tile = blockIdx.x * 8 + wave; tile < ntiles; tile += totalWaves) {
    // Prefetch next tile's A row (lowers to global_prefetch_b8).
    int tnext = tile + totalWaves;
    if (tnext < ntiles)
      __builtin_prefetch(A + (size_t)(tnext * 16 + mrow) * LDA, 0, 3);

    v8f acc[NT];
    #pragma unroll
    for (int nt = 0; nt < NT; ++nt)
      #pragma unroll
      for (int i = 0; i < 8; ++i) acc[nt][i] = 0.f;

    const uint32_t* Arow =
        (const uint32_t*)(A + (size_t)(tile * 16 + mrow) * LDA);

    #pragma unroll
    for (int kc = 0; kc < KC; ++kc) {
      union { uint32_t u[8]; v16bf v; } af;
      #pragma unroll
      for (int v = 0; v < 8; ++v) {
        int k = kc * 32 + ((v & 4) << 2) + h * 8 + ((v & 3) << 1);
        af.u[v] = Arow[k >> 1];
      }
      #pragma unroll
      for (int nt = 0; nt < NT; ++nt) {
        union { uint32_t u[8]; v16bf v; } bfr;
        const uint32_t* sp = &sB[((kc * NT + nt) * 32 + lane) * 8];
        #pragma unroll
        for (int v = 0; v < 8; ++v) bfr.u[v] = sp[v];
        acc[nt] = __builtin_amdgcn_wmma_f32_16x16x32_bf16(
            false, af.v, false, bfr.v, (short)0, acc[nt], false, false);
      }
    }

    // Bias, fp32 y store (immediate offsets off one base), stat accumulation.
    float* yp = Y + (size_t)(tile * 16 + 8 * h) * N + mrow;
    #pragma unroll
    for (int nt = 0; nt < NT; ++nt) {
      #pragma unroll
      for (int r = 0; r < 8; ++r) {
        float val = acc[nt][r] + bv[nt];
        yp[r * N + nt * 16] = val;
        ssum[nt] += val;
        ssq[nt]  += val * val;
      }
    }
  }

  // Deterministic in-block reduction of per-channel partials (fixed order).
  #pragma unroll
  for (int nt = 0; nt < NT; ++nt) {
    pSum[nt * 256 + tid] = ssum[nt];
    pSqr[nt * 256 + tid] = ssq[nt];
  }
  __syncthreads();
  if (tid < N) {
    int cl = tid & 15, ntv = tid >> 4;
    float s = 0.f, q = 0.f;
    for (int w = 0; w < 8; ++w)
      for (int hh = 0; hh < 2; ++hh) {
        int idx = ntv * 256 + w * 32 + hh * 16 + cl;
        s += pSum[idx]; q += pSqr[idx];
      }
    part[(size_t)blockIdx.x * (2 * N) + tid]     = s;
    part[(size_t)blockIdx.x * (2 * N) + N + tid] = q;
  }
}

// ---------------------------------------------------------------------------
// Kernel 4: fold block partials (fixed order => deterministic) into per-channel
// BN scale/shift: s = g*rsqrt(var+eps), t = beta - mean*s.
// ---------------------------------------------------------------------------
__global__ void bn_prep_kernel(const float* __restrict__ part, int nblocks, int N,
                               const float* __restrict__ g,
                               const float* __restrict__ beta,
                               float* __restrict__ s, float* __restrict__ t,
                               float count)
{
  int c = blockIdx.x * blockDim.x + threadIdx.x;
  if (c >= N) return;
  float sum = 0.f, sq = 0.f;
  for (int b = 0; b < nblocks; ++b) {
    sum += part[(size_t)b * (2 * N) + c];
    sq  += part[(size_t)b * (2 * N) + N + c];
  }
  float mean = sum / count;
  float var  = sq / count - mean * mean;
  float r    = rsqrtf(var + 1e-5f);
  float sc   = g[c] * r;
  s[c] = sc;
  t[c] = beta[c] - mean * sc;
}

// Kernel 5: y -> relu(y*s + t) as bf16 (input for the next GEMM). ld == 64.
__global__ void bn_relu_cast_kernel(const float* __restrict__ Y,
                                    const float* __restrict__ s,
                                    const float* __restrict__ t,
                                    unsigned short* __restrict__ Aout,
                                    size_t total)
{
  for (size_t i = (size_t)blockIdx.x * blockDim.x + threadIdx.x; i < total;
       i += (size_t)gridDim.x * blockDim.x) {
    int c = (int)(i & 63u);
    float v = fmaxf(fmaf(Y[i], s[c], t[c]), 0.f);
    Aout[i] = f2bf(v);
  }
}

// Kernel 6: final BN+ReLU on y3, max over K neighbors, write (B,128,M).
__global__ __launch_bounds__(128) void final_max_kernel(
    const float* __restrict__ Y, const float* __restrict__ s,
    const float* __restrict__ t, float* __restrict__ outp)
{
  int g = blockIdx.x;        // b*M + m
  int c = threadIdx.x;       // channel 0..127
  float sc = s[c], tc = t[c];
  float mx = -3.4e38f;
  const float* yr = Y + (size_t)g * KNN_ * 128 + c;
  #pragma unroll 4
  for (int j = 0; j < KNN_; ++j) {
    float v = fmaxf(fmaf(yr[(size_t)j * 128], sc, tc), 0.f);
    mx = fmaxf(mx, v);
  }
  int b = g / M_, m = g - b * M_;
  outp[CENT_OFF + ((size_t)b * 128 + c) * M_ + m] = mx;
}

// ---------------------------------------------------------------------------
extern "C" void kernel_launch(void* const* d_in, const int* in_sizes, int n_in,
                              void* d_out, int out_size, void* d_ws, size_t ws_size,
                              hipStream_t stream)
{
  (void)in_sizes; (void)n_in; (void)out_size; (void)ws_size;

  const float* xyz   = (const float*)d_in[0];
  const float* feats = (const float*)d_in[1];
  const float* w1    = (const float*)d_in[2];
  const float* b1    = (const float*)d_in[3];
  const float* g1    = (const float*)d_in[4];
  const float* be1   = (const float*)d_in[5];
  const float* w2    = (const float*)d_in[6];
  const float* b2    = (const float*)d_in[7];
  const float* g2    = (const float*)d_in[8];
  const float* be2   = (const float*)d_in[9];
  const float* w3    = (const float*)d_in[10];
  const float* b3    = (const float*)d_in[11];
  const float* g3    = (const float*)d_in[12];
  const float* be3   = (const float*)d_in[13];
  float* out = (float*)d_out;

  // ---- workspace layout (256B aligned) ----
  char* ws = (char*)d_ws;
  constexpr size_t OFF_CAT  = 0;                                  // L*96 bf16
  constexpr size_t SZ_CAT   = (size_t)L_ * LDA1 * 2;
  constexpr size_t OFF_ACT  = OFF_CAT + SZ_CAT;                   // L*64 bf16
  constexpr size_t SZ_ACT   = (size_t)L_ * 64 * 2;
  constexpr size_t OFF_Y    = OFF_ACT + SZ_ACT;                   // L*128 f32
  constexpr size_t SZ_Y     = (size_t)L_ * 128 * 4;
  constexpr size_t OFF_WP1  = OFF_Y + SZ_Y;                       // 3*4*256 u32
  constexpr size_t OFF_WP2  = OFF_WP1 + 3 * 4 * 256 * 4;
  constexpr size_t OFF_WP3  = OFF_WP2 + 2 * 4 * 256 * 4;
  constexpr size_t OFF_PART = OFF_WP3 + 2 * 8 * 256 * 4;          // 1024*256 f32
  constexpr size_t OFF_S1   = OFF_PART + (size_t)1024 * 256 * 4;
  constexpr size_t OFF_T1   = OFF_S1 + 512;
  constexpr size_t OFF_S2   = OFF_T1 + 512;
  constexpr size_t OFF_T2   = OFF_S2 + 512;
  constexpr size_t OFF_S3   = OFF_T2 + 512;
  constexpr size_t OFF_T3   = OFF_S3 + 512;

  unsigned short* cat = (unsigned short*)(ws + OFF_CAT);
  unsigned short* act = (unsigned short*)(ws + OFF_ACT);
  float*    y    = (float*)(ws + OFF_Y);
  uint32_t* wp1  = (uint32_t*)(ws + OFF_WP1);
  uint32_t* wp2  = (uint32_t*)(ws + OFF_WP2);
  uint32_t* wp3  = (uint32_t*)(ws + OFF_WP3);
  float*    part = (float*)(ws + OFF_PART);
  float *s1 = (float*)(ws + OFF_S1), *t1 = (float*)(ws + OFF_T1);
  float *s2 = (float*)(ws + OFF_S2), *t2 = (float*)(ws + OFF_T2);
  float *s3 = (float*)(ws + OFF_S3), *t3 = (float*)(ws + OFF_T3);

  const int ntiles  = L_ / 16;     // 32768
  const int gblocks = 1024;        // 8192 waves -> 4 tiles each
  const float cnt   = (float)L_;

  // 1) centers + KNN + gather
  knn_gather_kernel<<<B_ * M_, 256, 0, stream>>>(xyz, feats, cat, out);

  // 2) pack weights into WMMA B-fragment blobs
  pack_w_kernel<<<16, 256, 0, stream>>>(w1,  64, 67, 3, 4, wp1);
  pack_w_kernel<<<16, 256, 0, stream>>>(w2,  64, 64, 2, 4, wp2);
  pack_w_kernel<<<16, 256, 0, stream>>>(w3, 128, 64, 2, 8, wp3);

  // 3) layer 1: (L x 96) x (96 x 64)
  gemm_bn_kernel<3, 4><<<gblocks, 256, 0, stream>>>(cat, wp1, b1, y, part, ntiles);
  bn_prep_kernel<<<1, 64, 0, stream>>>(part, gblocks, 64, g1, be1, s1, t1, cnt);
  bn_relu_cast_kernel<<<4096, 256, 0, stream>>>(y, s1, t1, act, (size_t)L_ * 64);

  // 4) layer 2: (L x 64) x (64 x 64)
  gemm_bn_kernel<2, 4><<<gblocks, 256, 0, stream>>>(act, wp2, b2, y, part, ntiles);
  bn_prep_kernel<<<1, 64, 0, stream>>>(part, gblocks, 64, g2, be2, s2, t2, cnt);
  bn_relu_cast_kernel<<<4096, 256, 0, stream>>>(y, s2, t2, act, (size_t)L_ * 64);

  // 5) layer 3: (L x 64) x (64 x 128)
  gemm_bn_kernel<2, 8><<<gblocks, 256, 0, stream>>>(act, wp3, b3, y, part, ntiles);
  bn_prep_kernel<<<1, 128, 0, stream>>>(part, gblocks, 128, g3, be3, s3, t3, cnt);

  // 6) BN+ReLU+max over K, write (B,128,M) after centers
  final_max_kernel<<<B_ * M_, 128, 0, stream>>>(y, s3, t3, out);
}
